// MPConv_25099788877922
// MI455X (gfx1250) — matrix operand order repaired and checked by reading, not compiled, not run on hardware
//
#include <hip/hip_runtime.h>
#include <math.h>

typedef __attribute__((ext_vector_type(16))) __bf16 v16bf;
typedef __attribute__((ext_vector_type(8)))  __bf16 v8bf;
typedef __attribute__((ext_vector_type(8)))  float  v8f;

#define IN_CH    64
#define OUT_CH   64
#define EDGE_DIM 32
#define CAT_DIM  160   // 2*IN_CH + EDGE_DIM
#define WAVES_PER_BLOCK 8
#define THREADS (WAVES_PER_BLOCK * 32)

// ---------------------------------------------------------------------------
// Zero-fill output (d_out is poisoned before timing; scatter uses atomics).
// ---------------------------------------------------------------------------
__global__ void zero_out_kernel(float* __restrict__ out, int n) {
    int i = blockIdx.x * blockDim.x + threadIdx.x;
    int stride = gridDim.x * blockDim.x;
    for (; i < n; i += stride) out[i] = 0.0f;
}

// ---------------------------------------------------------------------------
// Fused edge-MLP message passing:
//   feats = [x[i], x[j], edge_attr]            (160 per edge)
//   h  = gelu(layernorm(feats @ W1 + b1))      (64)
//   m  = h @ W2 + b2                           (64)
//   out[j] += m   (atomic scatter-add)
//
// One wave processes a 16-edge tile. Both GEMMs run on v_wmma_f32_16x16x32_bf16
// with f32 accumulation. W1/W2 are staged once per block into LDS, pre-swizzled
// into WMMA B-fragment layout (lane = output column, K striped per ISA 7.12.2).
// ---------------------------------------------------------------------------
__launch_bounds__(THREADS)
__global__ void mpconv_edge_kernel(const float*     __restrict__ x,
                                   const long long* __restrict__ eidx,   // [2, E] int64
                                   const float*     __restrict__ eattr,  // [E, 32]
                                   const float*     __restrict__ W1,     // [160, 64]
                                   const float*     __restrict__ b1,     // [64]
                                   const float*     __restrict__ gma,    // [64]
                                   const float*     __restrict__ bta,    // [64]
                                   const float*     __restrict__ W2,     // [64, 64]
                                   const float*     __restrict__ b2,     // [64]
                                   float*           __restrict__ out,    // [N, 64]
                                   int E) {
    // B-fragment LDS staging: fragment index = k_chunk*4 + n_tile, 32 lanes x 16 bf16.
    __shared__ v16bf w1f[20][32];                 // 160x64 bf16 -> 20 KB
    __shared__ v16bf w2f[8][32];                  //  64x64 bf16 ->  8 KB
    __shared__ __bf16 hbuf[WAVES_PER_BLOCK][16 * 64];  // per-wave h transpose buffer, 16 KB

    const int tid = threadIdx.x;

    // ---- Preload W1 into B-fragment layout (lane = ((kl>>3)&1)*16 + n%16, elem = (kl&7)|((kl>>4)<<3)) ----
    for (int idx = tid; idx < CAT_DIM * OUT_CH; idx += THREADS) {
        int k = idx >> 6, n = idx & 63;
        int c = k >> 5, kl = k & 31;
        int nt = n >> 4, nlz = n & 15;
        int ln = (((kl >> 3) & 1) << 4) | nlz;
        ((__bf16*)&w1f[(c << 2) + nt][ln])[(kl & 7) | ((kl >> 4) << 3)] = (__bf16)W1[idx];
    }
    for (int idx = tid; idx < OUT_CH * OUT_CH; idx += THREADS) {
        int k = idx >> 6, n = idx & 63;
        int c = k >> 5, kl = k & 31;
        int nt = n >> 4, nlz = n & 15;
        int ln = (((kl >> 3) & 1) << 4) | nlz;
        ((__bf16*)&w2f[(c << 2) + nt][ln])[(kl & 7) | ((kl >> 4) << 3)] = (__bf16)W2[idx];
    }
    __syncthreads();

    const int wave = tid >> 5;
    const int lane = tid & 31;
    const int nl   = lane & 15;   // column index inside a 16-wide N tile / A-row id
    const int half = lane >> 4;   // which 16-lane half of the wave

    const int tile = blockIdx.x * WAVES_PER_BLOCK + wave;
    const int te   = tile << 4;   // first edge of this 16-edge tile
    if (te >= E) return;

    // ---- Per-lane gather pointers for A rows (lane pairs L and L+16 share row nl) ----
    int e = te + nl;
    if (e >= E) e = E - 1;        // safe clamp; scatter below is exact-guarded
    const long long ni = eidx[e];
    const long long nj = eidx[(long long)E + e];
    const float* xi = x + ni * IN_CH;
    const float* xj = x + nj * IN_CH;
    const float* ep = eattr + (long long)e * EDGE_DIM;

    // Per-lane column constants (N = nt*16 + nl)
    float bias1[4], bias2[4], g4[4], bt4[4];
#pragma unroll
    for (int nt = 0; nt < 4; ++nt) {
        bias1[nt] = b1[nt * 16 + nl];
        bias2[nt] = b2[nt * 16 + nl];
        g4[nt]    = gma[nt * 16 + nl];
        bt4[nt]   = bta[nt * 16 + nl];
    }

    // ======================= GEMM1: h = feats @ W1 + b1 =======================
    v8f acc[4];
#pragma unroll
    for (int nt = 0; nt < 4; ++nt)
#pragma unroll
        for (int v = 0; v < 8; ++v) acc[nt][v] = bias1[nt];

#pragma unroll
    for (int c = 0; c < 5; ++c) {
        // K-chunk c source: 0,1 -> x[i]; 2,3 -> x[j]; 4 -> edge_attr
        const float* s = (c == 0) ? xi
                       : (c == 1) ? (xi + 32)
                       : (c == 2) ? xj
                       : (c == 3) ? (xj + 32)
                       : ep;
        // A layout (ISA 7.12.2, 16-bit A 16x32): this lane holds two contiguous
        // 8-element K runs: [half*8, +8) and [16+half*8, +8).
        const float4 f0 = *(const float4*)(s + half * 8);
        const float4 f1 = *(const float4*)(s + half * 8 + 4);
        const float4 f2 = *(const float4*)(s + 16 + half * 8);
        const float4 f3 = *(const float4*)(s + 16 + half * 8 + 4);
        v16bf a;
        a[0]  = (__bf16)f0.x; a[1]  = (__bf16)f0.y; a[2]  = (__bf16)f0.z; a[3]  = (__bf16)f0.w;
        a[4]  = (__bf16)f1.x; a[5]  = (__bf16)f1.y; a[6]  = (__bf16)f1.z; a[7]  = (__bf16)f1.w;
        a[8]  = (__bf16)f2.x; a[9]  = (__bf16)f2.y; a[10] = (__bf16)f2.z; a[11] = (__bf16)f2.w;
        a[12] = (__bf16)f3.x; a[13] = (__bf16)f3.y; a[14] = (__bf16)f3.z; a[15] = (__bf16)f3.w;
#pragma unroll
        for (int nt = 0; nt < 4; ++nt) {
            v16bf b = w1f[(c << 2) + nt][lane];
            acc[nt] = __builtin_amdgcn_wmma_f32_16x16x32_bf16(
                false, a, false, b, (short)0, acc[nt], false, false);
        }
    }

    // ================== LayerNorm (per row, across 64 channels) ==================
    // C layout: VGPR v holds row M = v + half*8, lane nl holds channel nt*16+nl.
    // A full row lives in one 16-lane half -> xor-shuffle reduction over masks 1..8.
    float mu[8], rs[8];
#pragma unroll
    for (int v = 0; v < 8; ++v) {
        float s = acc[0][v] + acc[1][v] + acc[2][v] + acc[3][v];
        float q = acc[0][v] * acc[0][v] + acc[1][v] * acc[1][v]
                + acc[2][v] * acc[2][v] + acc[3][v] * acc[3][v];
#pragma unroll
        for (int m = 1; m <= 8; m <<= 1) {
            s += __shfl_xor(s, m, 32);
            q += __shfl_xor(q, m, 32);
        }
        float mean = s * (1.0f / 64.0f);
        float var  = q * (1.0f / 64.0f) - mean * mean;
        mu[v] = mean;
        rs[v] = __frsqrt_rn(var + 1e-5f);
    }

    // ============ GELU (exact) + transpose h into A layout via LDS ============
    __bf16* hb = &hbuf[wave][0];
#pragma unroll
    for (int v = 0; v < 8; ++v) {
        const int M = v + half * 8;
#pragma unroll
        for (int nt = 0; nt < 4; ++nt) {
            float hn = (acc[nt][v] - mu[v]) * rs[v] * g4[nt] + bt4[nt];
            float ge = 0.5f * hn * (1.0f + erff(hn * 0.70710678118654752f));
            hb[M * 64 + nt * 16 + nl] = (__bf16)ge;
        }
    }
    // Same-wave DS store -> DS load ordering (CDNA5 split counter).
    asm volatile("s_wait_dscnt 0" ::: "memory");

    // ======================= GEMM2: m = h @ W2 + b2 =======================
    v8f acc2[4];
#pragma unroll
    for (int nt = 0; nt < 4; ++nt)
#pragma unroll
        for (int v = 0; v < 8; ++v) acc2[nt][v] = bias2[nt];

#pragma unroll
    for (int c2 = 0; c2 < 2; ++c2) {
        const v8bf r0 = *(const v8bf*)&hb[nl * 64 + c2 * 32 + half * 8];
        const v8bf r1 = *(const v8bf*)&hb[nl * 64 + c2 * 32 + 16 + half * 8];
        v16bf a;
#pragma unroll
        for (int t = 0; t < 8; ++t) { a[t] = r0[t]; a[8 + t] = r1[t]; }
#pragma unroll
        for (int nt = 0; nt < 4; ++nt) {
            v16bf b = w2f[(c2 << 2) + nt][lane];
            acc2[nt] = __builtin_amdgcn_wmma_f32_16x16x32_bf16(
                false, a, false, b, (short)0, acc2[nt], false, false);
        }
    }

    // =================== Scatter-add messages onto target nodes ===================
#pragma unroll
    for (int v = 0; v < 8; ++v) {
        const int M  = v + half * 8;
        const int e2 = te + M;
        if (e2 < E) {
            const long long jn = eidx[(long long)E + e2];
            float* op = out + jn * OUT_CH;
#pragma unroll
            for (int nt = 0; nt < 4; ++nt)
                unsafeAtomicAdd(op + nt * 16 + nl, acc2[nt][v]);
        }
    }
}

// ---------------------------------------------------------------------------
// Launch
// ---------------------------------------------------------------------------
extern "C" void kernel_launch(void* const* d_in, const int* in_sizes, int n_in,
                              void* d_out, int out_size, void* d_ws, size_t ws_size,
                              hipStream_t stream) {
    const float*     x     = (const float*)d_in[0];
    const long long* eidx  = (const long long*)d_in[1];   // int64 edge_index [2, E]
    const float*     eattr = (const float*)d_in[2];
    const float*     W1    = (const float*)d_in[3];
    const float*     b1    = (const float*)d_in[4];
    const float*     gma   = (const float*)d_in[5];
    const float*     bta   = (const float*)d_in[6];
    const float*     W2    = (const float*)d_in[7];
    const float*     b2    = (const float*)d_in[8];
    float*           out   = (float*)d_out;

    const int E = in_sizes[1] / 2;   // edge_index has 2*E elements

    // Zero the (poisoned) output accumulator.
    zero_out_kernel<<<4096, 256, 0, stream>>>(out, out_size);

    const int tiles  = (E + 15) / 16;
    const int blocks = (tiles + WAVES_PER_BLOCK - 1) / WAVES_PER_BLOCK;
    mpconv_edge_kernel<<<blocks, THREADS, 0, stream>>>(
        x, eidx, eattr, W1, b1, gma, bta, W2, b2, out, E);
}
